// MatchNet_44203803411128
// MI455X (gfx1250) — compile-verified, matrix operand order checked by reading, not run on hardware
//
#include <hip/hip_runtime.h>
#include <math.h>

// ---------------- problem constants ----------------
#define BATCH 8
#define NPTS  2048
#define KNN   20
#define EMBD  512
#define BN_EPS 1e-5f
#define NEG_INF (-3.402823466e38f)

typedef __bf16 bf16;
typedef __attribute__((ext_vector_type(16))) __bf16 v16bf;
typedef __attribute__((ext_vector_type(8)))  __bf16 v8bf;
typedef __attribute__((ext_vector_type(8)))  float  v8f;

union V16 { v16bf v; v8bf h[2]; };

// =====================================================================
// Generic batched GEMM:  C[M,N] = A[M,K] * B[N,K]^T
// A, B bf16 row-major along K (K % 32 == 0), C fp32 row-major.
// Register-blocked: one 32x32 output tile per wave (4 WMMA accumulators,
// 2 A-fragments x 2 B-fragments per 32-deep K step -> 16 FLOP/byte from
// cache, double the intensity of a 16x16 tile).  8 waves per block.
// Emits v_wmma_f32_16x16x32_bf16.
// =====================================================================
__global__ void gemm_bt_wmma(const bf16* __restrict__ A,
                             const bf16* __restrict__ Bm,
                             float* __restrict__ C,
                             int M, int N, int Kd,
                             long strideA, long strideB, long strideC) {
  const int bz = blockIdx.z;
  A  += (long)bz * strideA;
  Bm += (long)bz * strideB;
  C  += (long)bz * strideC;

  const int lane = threadIdx.x & 31;
  const int wave = threadIdx.x >> 5;
  const int tilesN = N >> 5;                    // 32-wide N tiles
  const long nTiles = (long)(M >> 5) * tilesN;  // 32-tall M tiles
  const long tile = (long)blockIdx.x * 8 + wave;
  if (tile >= nTiles) return;

  const int tm = (int)(tile / tilesN) << 5;
  const int tn = (int)(tile % tilesN) << 5;

  // 16-bit A/B fragment layout: lane l holds row/col (l&15); K chunks
  // {0..7,16..23} for l<16 and {8..15,24..31} for l>=16 (per K-step of 32).
  const int khalf = (lane >> 4) << 3;           // 0 or 8
  const bf16* Arow0 = A  + (long)(tm + (lane & 15)) * Kd + khalf;
  const bf16* Arow1 = Arow0 + (long)16 * Kd;
  const bf16* Brow0 = Bm + (long)(tn + (lane & 15)) * Kd + khalf;
  const bf16* Brow1 = Brow0 + (long)16 * Kd;

  v8f acc00 = {}, acc01 = {}, acc10 = {}, acc11 = {};
  for (int kk = 0; kk < Kd; kk += 32) {
    V16 a0, a1, b0, b1;
    a0.h[0] = *(const v8bf*)(Arow0 + kk);
    a0.h[1] = *(const v8bf*)(Arow0 + kk + 16);
    a1.h[0] = *(const v8bf*)(Arow1 + kk);
    a1.h[1] = *(const v8bf*)(Arow1 + kk + 16);
    b0.h[0] = *(const v8bf*)(Brow0 + kk);
    b0.h[1] = *(const v8bf*)(Brow0 + kk + 16);
    b1.h[0] = *(const v8bf*)(Brow1 + kk);
    b1.h[1] = *(const v8bf*)(Brow1 + kk + 16);
    // CDNA5 speculative prefetch of upcoming K-lines (global_prefetch_b8)
    __builtin_prefetch(Arow0 + kk + 128, 0, 1);
    __builtin_prefetch(Brow0 + kk + 128, 0, 1);
    acc00 = __builtin_amdgcn_wmma_f32_16x16x32_bf16(
        false, a0.v, false, b0.v, (short)0, acc00, false, false);
    acc01 = __builtin_amdgcn_wmma_f32_16x16x32_bf16(
        false, a0.v, false, b1.v, (short)0, acc01, false, false);
    acc11 = __builtin_amdgcn_wmma_f32_16x16x32_bf16(
        false, a1.v, false, b1.v, (short)0, acc11, false, false);
    acc10 = __builtin_amdgcn_wmma_f32_16x16x32_bf16(
        false, a1.v, false, b0.v, (short)0, acc10, false, false);
  }

  // f32 C/D layout: VGPR j -> row base+j (lanes 0-15) / base+8+j (lanes 16-31)
  const int rsub = (lane >> 4) << 3;
  const int csub = lane & 15;
  float* C00 = C + (long)(tm + rsub) * N + tn + csub;
#pragma unroll
  for (int j = 0; j < 8; ++j) {
    C00[(long)j * N]            = acc00[j];
    C00[(long)j * N + 16]       = acc01[j];
    C00[(long)(j + 16) * N]     = acc10[j];
    C00[(long)(j + 16) * N + 16]= acc11[j];
  }
}

// =====================================================================
// fp32 -> bf16 conversion
// =====================================================================
__global__ void f32_to_bf16(const float* __restrict__ in, bf16* __restrict__ out, long n) {
  long i = (long)blockIdx.x * blockDim.x + threadIdx.x;
  if (i < n) out[i] = (bf16)in[i];
}

__global__ void zero_f(float* p, int n) {
  int i = blockIdx.x * blockDim.x + threadIdx.x;
  if (i < n) p[i] = 0.0f;
}

// =====================================================================
// kNN top-K (negative squared distance, ties -> lowest index).
// One block per (b, n). dist table + argmax reductions in LDS.
// =====================================================================
__global__ void knn_topk(const float* __restrict__ x, int* __restrict__ idx,
                         int N, int Kn) {
  __shared__ float dist[NPTS];
  __shared__ float rv[256];
  __shared__ int   ri[256];
  const long bn = blockIdx.x;
  const int n = (int)(bn % N), b = (int)(bn / N);
  const float cx = x[((long)b * 3 + 0) * N + n];
  const float cy = x[((long)b * 3 + 1) * N + n];
  const float cz = x[((long)b * 3 + 2) * N + n];
  for (int m = threadIdx.x; m < N; m += blockDim.x) {
    float dx = x[((long)b * 3 + 0) * N + m] - cx;
    float dy = x[((long)b * 3 + 1) * N + m] - cy;
    float dz = x[((long)b * 3 + 2) * N + m] - cz;
    dist[m] = -(dx * dx + dy * dy + dz * dz);
  }
  __syncthreads();
  for (int s = 0; s < Kn; ++s) {
    float bv = NEG_INF; int bi = 0;
    for (int m = threadIdx.x; m < N; m += blockDim.x) {
      float v = dist[m];
      if (v > bv) { bv = v; bi = m; }
    }
    rv[threadIdx.x] = bv; ri[threadIdx.x] = bi;
    __syncthreads();
    for (int off = 128; off > 0; off >>= 1) {
      if ((int)threadIdx.x < off) {
        float ov = rv[threadIdx.x + off]; int oi = ri[threadIdx.x + off];
        if (ov > rv[threadIdx.x] || (ov == rv[threadIdx.x] && oi < ri[threadIdx.x])) {
          rv[threadIdx.x] = ov; ri[threadIdx.x] = oi;
        }
      }
      __syncthreads();
    }
    if (threadIdx.x == 0) { idx[bn * Kn + s] = ri[0]; dist[ri[0]] = NEG_INF; }
    __syncthreads();
  }
}

// =====================================================================
// Edge feature + conv1 (C=6 -> O=64).  One block per (b,n,k), 64 threads.
// =====================================================================
__global__ void edge_conv1(const float* __restrict__ x, const int* __restrict__ idx,
                           const float* __restrict__ w1, float* __restrict__ y,
                           int N, int Kn) {
  __shared__ float e[6];
  const long blk = blockIdx.x;             // b*N*K + n*K + k
  const int k = (int)(blk % Kn);
  const long bn = blk / Kn;
  const int n = (int)(bn % N), b = (int)(bn / N);
  const int j = idx[bn * Kn + k];
  if (threadIdx.x < 3) {
    float c  = x[((long)b * 3 + threadIdx.x) * N + n];
    float nb = x[((long)b * 3 + threadIdx.x) * N + j];
    e[threadIdx.x]     = nb - c;   // nbr - ctr
    e[3 + threadIdx.x] = c;        // ctr
  }
  __syncthreads();
  const int o = threadIdx.x;
  float acc = 0.0f;
#pragma unroll
  for (int c = 0; c < 6; ++c) acc += w1[o * 6 + c] * e[c];
  y[(bn * Kn + k) * 64 + o] = acc;
}

// =====================================================================
// Global BN statistics: per-channel sum / sumsq via LDS float atomics
// (ds_add_f32) + one global atomic per channel per block.
// =====================================================================
__global__ void bn_stats(const float* __restrict__ y, float* __restrict__ stats,
                         long R, int C) {
  extern __shared__ float sh[];            // 2*C floats
  for (int c = threadIdx.x; c < 2 * C; c += blockDim.x) sh[c] = 0.0f;
  __syncthreads();
  const long total = R * (long)C;
  for (long i = (long)blockIdx.x * blockDim.x + threadIdx.x; i < total;
       i += (long)gridDim.x * blockDim.x) {
    float v = y[i];
    int c = (int)(i % C);
    atomicAdd(&sh[c], v);
    atomicAdd(&sh[C + c], v * v);
  }
  __syncthreads();
  for (int c = threadIdx.x; c < 2 * C; c += blockDim.x) atomicAdd(&stats[c], sh[c]);
}

// =====================================================================
// BN (global mean/var) + ReLU + bf16 convert.
// =====================================================================
__global__ void bn_relu_bf16(const float* __restrict__ y, const float* __restrict__ stats,
                             const float* __restrict__ g, const float* __restrict__ bta,
                             bf16* __restrict__ out, long R, int C, float invCnt) {
  long i = (long)blockIdx.x * blockDim.x + threadIdx.x;
  if (i >= R * (long)C) return;
  int c = (int)(i % C);
  float mean = stats[c] * invCnt;
  float var  = stats[C + c] * invCnt - mean * mean;
  float v = (y[i] - mean) * rsqrtf(var + BN_EPS) * g[c] + bta[c];
  out[i] = (bf16)fmaxf(v, 0.0f);
}

// =====================================================================
// max over K neighbors -> write channel slice into the 512-wide cat buffer
// =====================================================================
__global__ void maxk_to_cat(const bf16* __restrict__ h, bf16* __restrict__ cat,
                            long BN, int Kn, int C, int chOff) {
  long i = (long)blockIdx.x * blockDim.x + threadIdx.x;
  if (i >= BN * (long)C) return;
  int c = (int)(i % C);
  long bn = i / C;
  float m = NEG_INF;
  for (int k = 0; k < Kn; ++k)
    m = fmaxf(m, (float)h[(bn * Kn + k) * (long)C + c]);
  cat[bn * EMBD + chOff + c] = (bf16)m;
}

// =====================================================================
// Row softmax (in place) + argmax + entropy weight w = -1/(sum p log(p+eps)+eps)
// One 256-thread block per row of the (B*N, M) score matrix.
// =====================================================================
__global__ void softmax_row(float* __restrict__ sc, float* __restrict__ wout,
                            int* __restrict__ amax, int Mcols) {
  __shared__ float rv[256];
  __shared__ int   ri[256];
  __shared__ float bc[2];
  const long row = blockIdx.x;
  float* p = sc + row * (long)Mcols;
  const int tid = threadIdx.x;

  // pass 1: max + argmax
  float bv = NEG_INF; int bi = 0;
  for (int m = tid; m < Mcols; m += 256) {
    float v = p[m];
    if (v > bv) { bv = v; bi = m; }
  }
  rv[tid] = bv; ri[tid] = bi;
  __syncthreads();
  for (int off = 128; off > 0; off >>= 1) {
    if (tid < off) {
      float ov = rv[tid + off]; int oi = ri[tid + off];
      if (ov > rv[tid] || (ov == rv[tid] && oi < ri[tid])) { rv[tid] = ov; ri[tid] = oi; }
    }
    __syncthreads();
  }
  if (tid == 0) { bc[0] = rv[0]; amax[row] = ri[0]; }
  __syncthreads();
  const float mx = bc[0];

  // pass 2: sum of exp
  float s = 0.0f;
  for (int m = tid; m < Mcols; m += 256) s += expf(p[m] - mx);
  __syncthreads();
  rv[tid] = s; __syncthreads();
  for (int off = 128; off > 0; off >>= 1) {
    if (tid < off) rv[tid] += rv[tid + off];
    __syncthreads();
  }
  if (tid == 0) bc[1] = rv[0];
  __syncthreads();
  const float inv = 1.0f / bc[1];

  // pass 3: normalize + entropy
  float ent = 0.0f;
  for (int m = tid; m < Mcols; m += 256) {
    float pv = expf(p[m] - mx) * inv;
    p[m] = pv;
    ent += pv * logf(pv + 1e-8f);
  }
  __syncthreads();
  rv[tid] = ent; __syncthreads();
  for (int off = 128; off > 0; off >>= 1) {
    if (tid < off) rv[tid] += rv[tid + off];
    __syncthreads();
  }
  if (tid == 0) wout[row] = -1.0f / (rv[0] + 1e-8f);
}

// =====================================================================
// Weighted Procrustes head: centroids, H = sum wn * s * c^T - sc*tc^T,
// 3x3 SVD (Jacobi on H^T H), det-corrected R, t.  One block per batch.
// =====================================================================
__device__ static float det3(const float m[3][3]) {
  return m[0][0] * (m[1][1] * m[2][2] - m[1][2] * m[2][1])
       - m[0][1] * (m[1][0] * m[2][2] - m[1][2] * m[2][0])
       + m[0][2] * (m[1][0] * m[2][1] - m[1][1] * m[2][0]);
}

__global__ void svd_head(const float* __restrict__ src, const float* __restrict__ tgt,
                         const float* __restrict__ wrow, const int* __restrict__ amax,
                         float* __restrict__ Rout, float* __restrict__ tout, int N) {
  __shared__ float sh[256];
  __shared__ float tot[16];
  const int b = blockIdx.x;
  const int tid = threadIdx.x;

  // phase 1: sum of w
  float sw = 0.0f;
  for (int n = tid; n < N; n += 256) sw += wrow[(long)b * N + n];
  sh[tid] = sw; __syncthreads();
  for (int off = 128; off > 0; off >>= 1) {
    if (tid < off) sh[tid] += sh[tid + off];
    __syncthreads();
  }
  if (tid == 0) tot[15] = sh[0];
  __syncthreads();
  const float invSW = 1.0f / (tot[15] + 1e-8f);

  // phase 2: S1 = sum wn*s, S2 = sum wn*c, M = sum wn*s*c^T
  float part[15];
#pragma unroll
  for (int i = 0; i < 15; ++i) part[i] = 0.0f;
  for (int n = tid; n < N; n += 256) {
    float wn = wrow[(long)b * N + n] * invSW;
    float sx = src[((long)b * 3 + 0) * N + n];
    float sy = src[((long)b * 3 + 1) * N + n];
    float sz = src[((long)b * 3 + 2) * N + n];
    int j = amax[(long)b * N + n];
    float tx = tgt[((long)b * 3 + 0) * N + j];
    float ty = tgt[((long)b * 3 + 1) * N + j];
    float tz = tgt[((long)b * 3 + 2) * N + j];
    part[0] += wn * sx; part[1] += wn * sy; part[2] += wn * sz;
    part[3] += wn * tx; part[4] += wn * ty; part[5] += wn * tz;
    part[6]  += wn * sx * tx; part[7]  += wn * sx * ty; part[8]  += wn * sx * tz;
    part[9]  += wn * sy * tx; part[10] += wn * sy * ty; part[11] += wn * sy * tz;
    part[12] += wn * sz * tx; part[13] += wn * sz * ty; part[14] += wn * sz * tz;
  }
  for (int comp = 0; comp < 15; ++comp) {
    __syncthreads();
    sh[tid] = part[comp]; __syncthreads();
    for (int off = 128; off > 0; off >>= 1) {
      if (tid < off) sh[tid] += sh[tid + off];
      __syncthreads();
    }
    if (tid == 0) tot[comp] = sh[0];
  }
  __syncthreads();
  if (tid != 0) return;

  float sc[3] = { tot[0], tot[1], tot[2] };
  float tc[3] = { tot[3], tot[4], tot[5] };
  float H[3][3];
  for (int i = 0; i < 3; ++i)
    for (int j2 = 0; j2 < 3; ++j2)
      H[i][j2] = tot[6 + i * 3 + j2] - sc[i] * tc[j2] + (i == j2 ? 1e-8f : 0.0f);

  // A = H^T H (symmetric), Jacobi eigendecomposition -> V, eigenvalues
  float A[3][3], V[3][3];
  for (int i = 0; i < 3; ++i)
    for (int j2 = 0; j2 < 3; ++j2) {
      float acc = 0.0f;
      for (int k = 0; k < 3; ++k) acc += H[k][i] * H[k][j2];
      A[i][j2] = acc;
      V[i][j2] = (i == j2) ? 1.0f : 0.0f;
    }
  for (int sweep = 0; sweep < 12; ++sweep) {
    for (int pq = 0; pq < 3; ++pq) {
      int p = (pq == 2) ? 1 : 0;
      int q = (pq == 0) ? 1 : 2;
      float apq = A[p][q];
      if (fabsf(apq) < 1e-14f) continue;
      float theta = (A[q][q] - A[p][p]) / (2.0f * apq);
      float t = (theta >= 0.0f ? 1.0f : -1.0f) / (fabsf(theta) + sqrtf(theta * theta + 1.0f));
      float c = rsqrtf(t * t + 1.0f);
      float s = t * c;
      for (int k = 0; k < 3; ++k) {
        float akp = A[k][p], akq = A[k][q];
        A[k][p] = c * akp - s * akq;
        A[k][q] = s * akp + c * akq;
      }
      for (int k = 0; k < 3; ++k) {
        float apk = A[p][k], aqk = A[q][k];
        A[p][k] = c * apk - s * aqk;
        A[q][k] = s * apk + c * aqk;
      }
      for (int k = 0; k < 3; ++k) {
        float vkp = V[k][p], vkq = V[k][q];
        V[k][p] = c * vkp - s * vkq;
        V[k][q] = s * vkp + c * vkq;
      }
    }
  }
  // sort eigenpairs descending
  float lam[3] = { A[0][0], A[1][1], A[2][2] };
  for (int i = 0; i < 2; ++i)
    for (int j2 = i + 1; j2 < 3; ++j2)
      if (lam[j2] > lam[i]) {
        float tl = lam[i]; lam[i] = lam[j2]; lam[j2] = tl;
        for (int k = 0; k < 3; ++k) { float tv = V[k][i]; V[k][i] = V[k][j2]; V[k][j2] = tv; }
      }
  // U columns = H * v_i / s_i
  float U[3][3];
  for (int i = 0; i < 3; ++i) {
    float si = sqrtf(fmaxf(lam[i], 0.0f));
    float inv = 1.0f / fmaxf(si, 1e-12f);
    for (int r = 0; r < 3; ++r) {
      float acc = 0.0f;
      for (int k = 0; k < 3; ++k) acc += H[r][k] * V[k][i];
      U[r][i] = acc * inv;
    }
  }
  // det(V U^T)
  float VUt[3][3];
  for (int i = 0; i < 3; ++i)
    for (int j2 = 0; j2 < 3; ++j2) {
      float acc = 0.0f;
      for (int k = 0; k < 3; ++k) acc += V[i][k] * U[j2][k];
      VUt[i][j2] = acc;
    }
  float d = det3(VUt);
  // R = V diag(1,1,d) U^T
  float R[3][3];
  float Dg[3] = { 1.0f, 1.0f, d };
  for (int i = 0; i < 3; ++i)
    for (int j2 = 0; j2 < 3; ++j2) {
      float acc = 0.0f;
      for (int k = 0; k < 3; ++k) acc += V[i][k] * Dg[k] * U[j2][k];
      R[i][j2] = acc;
    }
  // t = -R*sc + tc
  for (int i = 0; i < 3; ++i) {
    float acc = 0.0f;
    for (int k = 0; k < 3; ++k) acc += R[i][k] * sc[k];
    tout[b * 3 + i] = tc[i] - acc;
    for (int j2 = 0; j2 < 3; ++j2) Rout[b * 9 + i * 3 + j2] = R[i][j2];
  }
}

// =====================================================================
// Host-side orchestration
// =====================================================================
extern "C" void kernel_launch(void* const* d_in, const int* in_sizes, int n_in,
                              void* d_out, int out_size, void* d_ws, size_t ws_size,
                              hipStream_t stream) {
  (void)in_sizes; (void)n_in; (void)out_size; (void)ws_size;
  const int B = BATCH, N = NPTS, Kn = KNN;
  const long BNK = (long)B * N * Kn;        // 327680
  const long BN  = (long)B * N;             // 16384

  const float* src = (const float*)d_in[0];
  const float* tgt = (const float*)d_in[1];
  // d_in[2] = temp (unused by reference math), d_in[33] = i (unused)
  // params: insertion order w1,g1,b1,...,w5,g5,b5 ; src at 3, tgt at 18
  auto P = [&](int base, int layer, int which) -> const float* {
    return (const float*)d_in[base + (layer - 1) * 3 + which];  // 0=w,1=g,2=b
  };

  // ---- workspace carve-up (256B aligned) ----
  size_t off = 0;
  auto alloc = [&](size_t bytes) -> char* {
    off = (off + 255) & ~(size_t)255;
    char* p = (char*)d_ws + off;
    off += bytes;
    return p;
  };
  const int wElems[4] = { 64 * 64, 128 * 64, 256 * 128, 512 * 512 };
  bf16* wbf[2][4];
  for (int pl = 0; pl < 2; ++pl)
    for (int l = 0; l < 4; ++l) wbf[pl][l] = (bf16*)alloc((size_t)wElems[l] * 2);
  int*   idxBuf[2]; idxBuf[0] = (int*)alloc(BNK * 4); idxBuf[1] = (int*)alloc(BNK * 4);
  float* y    = (float*)alloc((size_t)BNK * 256 * 4);
  bf16*  hA   = (bf16*)alloc((size_t)BNK * 256 * 2);
  bf16*  hB   = (bf16*)alloc((size_t)BNK * 256 * 2);
  bf16*  cat  = (bf16*)alloc((size_t)BN * EMBD * 2);
  bf16*  emb[2]; emb[0] = (bf16*)alloc((size_t)BN * EMBD * 2); emb[1] = (bf16*)alloc((size_t)BN * EMBD * 2);
  float* stats = (float*)alloc(2 * EMBD * 4);
  float* wrowB = (float*)alloc((size_t)BN * 4);
  int*   amaxB = (int*)alloc((size_t)BN * 4);

  float* Rout   = (float*)d_out;            // (8,3,3)
  float* tout   = Rout + B * 9;             // (8,3)
  float* scores = tout + B * 3;             // (8,2048,2048)

  auto launch_gemm = [&](const bf16* A, const bf16* Bm, float* C,
                         int M, int Nn, int Kd, long sA, long sB, long sC, int bz) {
    long tiles = ((long)(M >> 5) * (Nn >> 5) + 7) / 8;   // 32x32 tiles, 8 waves/block
    dim3 grid((unsigned)tiles, 1, (unsigned)bz);
    gemm_bt_wmma<<<grid, 256, 0, stream>>>(A, Bm, C, M, Nn, Kd, sA, sB, sC);
  };
  auto launch_stats = [&](const float* yy, long R, int C) {
    zero_f<<<(2 * C + 255) / 256, 256, 0, stream>>>(stats, 2 * C);
    bn_stats<<<1024, 256, (size_t)2 * C * 4, stream>>>(yy, stats, R, C);
  };

  // ---- DGCNN pipelines: pl=0 -> src (params_src), pl=1 -> tgt (params_tgt)
  for (int pl = 0; pl < 2; ++pl) {
    const float* x = (pl == 0) ? src : tgt;
    const int base = (pl == 0) ? 3 : 18;

    // convert weights w2..w5 to bf16 once
    for (int l = 0; l < 4; ++l) {
      long ne = wElems[l];
      f32_to_bf16<<<(unsigned)((ne + 255) / 256), 256, 0, stream>>>(P(base, l + 2, 0), wbf[pl][l], ne);
    }

    // kNN graph
    knn_topk<<<(unsigned)BN, 256, 0, stream>>>(x, idxBuf[pl], N, Kn);

    // conv1: edge features (C=6) -> 64, scalar
    edge_conv1<<<(unsigned)BNK, 64, 0, stream>>>(x, idxBuf[pl], P(base, 1, 0), y, N, Kn);
    launch_stats(y, BNK, 64);
    bn_relu_bf16<<<(unsigned)((BNK * 64 + 255) / 256), 256, 0, stream>>>(
        y, stats, P(base, 1, 1), P(base, 1, 2), hA, BNK, 64, 1.0f / (float)BNK);
    maxk_to_cat<<<(unsigned)((BN * 64 + 255) / 256), 256, 0, stream>>>(hA, cat, BN, Kn, 64, 0);

    // conv2: 64 -> 64
    launch_gemm(hA, wbf[pl][0], y, (int)BNK, 64, 64, 0, 0, 0, 1);
    launch_stats(y, BNK, 64);
    bn_relu_bf16<<<(unsigned)((BNK * 64 + 255) / 256), 256, 0, stream>>>(
        y, stats, P(base, 2, 1), P(base, 2, 2), hB, BNK, 64, 1.0f / (float)BNK);
    maxk_to_cat<<<(unsigned)((BN * 64 + 255) / 256), 256, 0, stream>>>(hB, cat, BN, Kn, 64, 64);

    // conv3: 64 -> 128
    launch_gemm(hB, wbf[pl][1], y, (int)BNK, 128, 64, 0, 0, 0, 1);
    launch_stats(y, BNK, 128);
    bn_relu_bf16<<<(unsigned)((BNK * 128 + 255) / 256), 256, 0, stream>>>(
        y, stats, P(base, 3, 1), P(base, 3, 2), hA, BNK, 128, 1.0f / (float)BNK);
    maxk_to_cat<<<(unsigned)((BN * 128 + 255) / 256), 256, 0, stream>>>(hA, cat, BN, Kn, 128, 128);

    // conv4: 128 -> 256
    launch_gemm(hA, wbf[pl][2], y, (int)BNK, 256, 128, 0, 0, 0, 1);
    launch_stats(y, BNK, 256);
    bn_relu_bf16<<<(unsigned)((BNK * 256 + 255) / 256), 256, 0, stream>>>(
        y, stats, P(base, 4, 1), P(base, 4, 2), hB, BNK, 256, 1.0f / (float)BNK);
    maxk_to_cat<<<(unsigned)((BN * 256 + 255) / 256), 256, 0, stream>>>(hB, cat, BN, Kn, 256, 256);

    // conv5: 512 -> 512 over the concatenated max-pooled features
    launch_gemm(cat, wbf[pl][3], y, (int)BN, EMBD, EMBD, 0, 0, 0, 1);
    launch_stats(y, BN, EMBD);
    bn_relu_bf16<<<(unsigned)((BN * EMBD + 255) / 256), 256, 0, stream>>>(
        y, stats, P(base, 5, 1), P(base, 5, 2), emb[pl], BN, EMBD, 1.0f / (float)BN);
  }

  // ---- scores GEMM (batched): s[b,n,m] = <src_emb[b,n,:], tgt_emb[b,m,:]>
  launch_gemm(emb[0], emb[1], scores, N, N, EMBD,
              (long)N * EMBD, (long)N * EMBD, (long)N * N, B);

  // ---- softmax + argmax + entropy weights (in place on d_out scores)
  softmax_row<<<(unsigned)BN, 256, 0, stream>>>(scores, wrowB, amaxB, N);

  // ---- weighted SVD head -> R, t
  svd_head<<<B, 256, 0, stream>>>(src, tgt, wrowB, amaxB, Rout, tout, N);
}
